// SSRGRUModel_30116310680357
// MI455X (gfx1250) — matrix operand order repaired and checked by reading, not compiled, not run on hardware
//
#include <hip/hip_runtime.h>
#include <hip/hip_bf16.h>
#include <math.h>
#include <stdint.h>

// ---------------------------------------------------------------------------
// Problem dimensions (fixed by the reference)
// ---------------------------------------------------------------------------
#define S_LEN 512
#define BATCH 128
#define IN0   256
#define HID   512
#define G3    (3 * HID)     // 1536
#define CHUNK 64            // gx precompute chunk (steps)

typedef __attribute__((ext_vector_type(16))) __bf16 v16bf;
typedef __attribute__((ext_vector_type(8)))  __bf16 v8bf;
typedef __attribute__((ext_vector_type(8)))  float  v8f;

static __device__ __forceinline__ float sigmoidf_(float x) {
    return 1.0f / (1.0f + __expf(-x));
}

// ---------------------------------------------------------------------------
// gfx1250 async global->LDS staging (ASYNCcnt path), with safe fallback
// ---------------------------------------------------------------------------
#if __has_builtin(__builtin_amdgcn_global_load_async_to_lds_b128) && \
    __has_builtin(__builtin_amdgcn_s_wait_asynccnt)
#define USE_ASYNC_LDS 1
#else
#define USE_ASYNC_LDS 0
#endif

// The builtin expects: (addrspace(1) v4i*, addrspace(3) v4i*, imm offset, imm cpol)
// where v4i = int __attribute__((vector_size(16))) (per hipcc diagnostic).
typedef int vsi4 __attribute__((vector_size(16)));
typedef __attribute__((address_space(1))) vsi4 as1_vsi4;
typedef __attribute__((address_space(3))) vsi4 as3_vsi4;

// copy 16B (8 bf16) from global to LDS, per lane
static __device__ __forceinline__ void stage16(const __bf16* g, __bf16* l) {
#if USE_ASYNC_LDS
    __builtin_amdgcn_global_load_async_to_lds_b128(
        (as1_vsi4*)g, (as3_vsi4*)l, 0, 0);
#else
    *(v8bf*)l = *(const v8bf*)g;
#endif
}

static __device__ __forceinline__ void stage_fence() {
#if USE_ASYNC_LDS
    __builtin_amdgcn_s_wait_asynccnt(0);
#endif
    __syncthreads();
}

// ---------------------------------------------------------------------------
// WMMA helpers (CDNA5 wave32 layouts, CDNA5 ISA §7.12.2)
// ---------------------------------------------------------------------------
// A fragment, 16x32 bf16, row-major source (row = lane&15):
//   elements j=0..7  -> K = k0 + 8*halfId + j
//   elements j=8..15 -> K = k0 + 16 + 8*halfId + (j-8)
static __device__ __forceinline__ v16bf load_a_frag(const __bf16* __restrict__ row,
                                                    int k0, int halfId) {
    v8bf lo = *(const v8bf*)(row + k0 + 8 * halfId);
    v8bf hi = *(const v8bf*)(row + k0 + 16 + 8 * halfId);
    return __builtin_shufflevector(lo, hi, 0, 1, 2, 3, 4, 5, 6, 7,
                                           8, 9, 10, 11, 12, 13, 14, 15);
}

// B fragment, 32x16 bf16 (K x N): lane holds column n = lane&15,
// elements j=0..7 -> K = k0 + 8*halfId + j, j=8..15 -> K = k0 + 16 + 8*halfId + (j-8).
// `rowbase` points at the start of column n's weight row (LDS or global).
static __device__ __forceinline__ v16bf load_b_row(const __bf16* rowbase,
                                                   int k0, int halfId) {
    v8bf lo = *(const v8bf*)(rowbase + k0 + 8 * halfId);
    v8bf hi = *(const v8bf*)(rowbase + k0 + 16 + 8 * halfId);
    return __builtin_shufflevector(lo, hi, 0, 1, 2, 3, 4, 5, 6, 7,
                                           8, 9, 10, 11, 12, 13, 14, 15);
}

static __device__ __forceinline__ v8f wmma_bf16(v16bf a, v16bf b, v8f c) {
    return __builtin_amdgcn_wmma_f32_16x16x32_bf16(false, a, false, b,
                                                   (short)0, c, false, false);
}

// ---------------------------------------------------------------------------
// Elementwise utility kernels
// ---------------------------------------------------------------------------
__global__ void k_f2bf(const float* __restrict__ src, __bf16* __restrict__ dst, size_t n) {
    size_t i = (size_t)blockIdx.x * blockDim.x + threadIdx.x;
    if (i < n) dst[i] = (__bf16)src[i];
}

__global__ void k_zero_f32(float* __restrict__ p, size_t n) {
    size_t i = (size_t)blockIdx.x * blockDim.x + threadIdx.x;
    if (i < n) p[i] = 0.0f;
}

__global__ void k_zero_bf(__bf16* __restrict__ p, size_t n) {
    size_t i = (size_t)blockIdx.x * blockDim.x + threadIdx.x;
    if (i < n) p[i] = (__bf16)0.0f;
}

__global__ void k_copy_f32(const float* __restrict__ s, float* __restrict__ d, size_t n) {
    size_t i = (size_t)blockIdx.x * blockDim.x + threadIdx.x;
    if (i < n) d[i] = s[i];
}

// ---------------------------------------------------------------------------
// Batched input-projection GEMM:  C[M,N] = A[M,K](bf16) * W[N,K]^T(bf16) + bias
// Block = 128(M) x 64(N): 8 waves stacked in M share a 64-row B tile staged
// in LDS (row stride K+8 bf16 -> every ds_load_b128 spreads over all banks).
// ---------------------------------------------------------------------------
#define GEMM_LDS_ROWS 64
#define GEMM_LDS_RS   (512 + 8)   // max K + pad, in bf16 elements

__global__ __launch_bounds__(256, 1) void k_gemm_bias(
    const __bf16* __restrict__ A, const __bf16* __restrict__ W,
    const float* __restrict__ bias, float* __restrict__ C,
    int M, int N, int K, int kChunksLog2)
{
    __shared__ __bf16 smem[GEMM_LDS_ROWS * GEMM_LDS_RS];

    const int tid    = threadIdx.x;
    const int lane   = tid & 31;
    const int wave   = tid >> 5;
    const int halfId = lane >> 4;
    const int nlan   = lane & 15;

    const int nStrips = N >> 6;                   // 64-col strips
    const int bm = blockIdx.x / nStrips;
    const int bn = blockIdx.x % nStrips;
    const int m0 = bm * 128 + wave * 16;
    const int n0 = bn * 64;
    const int rs = K + 8;                         // LDS row stride (bf16)

    // --- cooperative stage of B tile: 64 rows x K bf16 ---
    {
        const int kc    = 1 << kChunksLog2;       // 16B chunks per row
        const int total = GEMM_LDS_ROWS << kChunksLog2;
        for (int i = tid; i < total; i += 256) {
            const int row = i >> kChunksLog2;
            const int c   = (i & (kc - 1)) * 8;
            stage16(W + (size_t)(n0 + row) * K + c, smem + row * rs + c);
        }
    }
    stage_fence();

    v8f acc0 = {}, acc1 = {}, acc2 = {}, acc3 = {};
    const __bf16* arow = A + (size_t)(m0 + nlan) * K;
    const __bf16* b0r = smem + ( 0 + nlan) * rs;
    const __bf16* b1r = smem + (16 + nlan) * rs;
    const __bf16* b2r = smem + (32 + nlan) * rs;
    const __bf16* b3r = smem + (48 + nlan) * rs;

    for (int k0 = 0; k0 < K; k0 += 32) {
        v16bf a  = load_a_frag(arow, k0, halfId);
        v16bf b0 = load_b_row(b0r, k0, halfId);
        v16bf b1 = load_b_row(b1r, k0, halfId);
        v16bf b2 = load_b_row(b2r, k0, halfId);
        v16bf b3 = load_b_row(b3r, k0, halfId);
        acc0 = wmma_bf16(a, b0, acc0);
        acc1 = wmma_bf16(a, b1, acc1);
        acc2 = wmma_bf16(a, b2, acc2);
        acc3 = wmma_bf16(a, b3, acc3);
    }

    const float bi0 = bias[n0 +  0 + nlan];
    const float bi1 = bias[n0 + 16 + nlan];
    const float bi2 = bias[n0 + 32 + nlan];
    const float bi3 = bias[n0 + 48 + nlan];
#pragma unroll
    for (int r = 0; r < 8; ++r) {
        int mrow = m0 + r + 8 * halfId;           // C/D layout: M = r + 8*(lane>=16)
        float* crow = C + (size_t)mrow * N + n0;
        crow[ 0 + nlan] = acc0[r] + bi0;
        crow[16 + nlan] = acc1[r] + bi1;
        crow[32 + nlan] = acc2[r] + bi2;
        crow[48 + nlan] = acc3[r] + bi3;
    }
}

// ---------------------------------------------------------------------------
// Fused recurrent cell eval.  One wave = one 16x16 (B x H) tile; the block's
// shared Wh tile (3 gates x 16 rows x 512 K = 48KB) is staged in LDS once and
// reused by all 8 waves.  Epilogue fuses gates + RK2 update:
//   phase 0:  f1 = (1-ig)*(ng - h);  bfOut = bf16(h + sb*f1)        (= half)
//   phase 1:  f2 from half;  hn = h + (1-b21)*f1 + b21*f2
// ---------------------------------------------------------------------------
#define CELL_LDS_RS (HID + 8)     // 520 bf16 per row

__global__ __launch_bounds__(256, 1) void k_cell(
    const __bf16* __restrict__ hbfIn,   // [B,H] bf16 GEMM A (h or half)
    const __bf16* __restrict__ Whbf,    // [3H,H] bf16
    const float*  __restrict__ bh,      // [3H]
    const float*  __restrict__ gxs,     // [B,3H] precomputed x-projection for this step
    const float*  __restrict__ hcur,    // [B,H] fp32 hidden at step start
    const float*  __restrict__ betaPtr, // scalar
    float*        __restrict__ f1,      // [B,H] phase0: write, phase1: read
    __bf16*       __restrict__ bfOut,   // phase0: halfbf; phase1: next-step bf16 hidden
    float*        __restrict__ fOut,    // phase1: next-step fp32 hidden (unused phase0)
    float*        __restrict__ outF32,  // phase1 optional extra fp32 output (layer 1)
    int phase)
{
    __shared__ __bf16 smem[48 * CELL_LDS_RS];

    const int tid    = threadIdx.x;
    const int lane   = tid & 31;
    const int wave   = tid >> 5;
    const int halfId = lane >> 4;
    const int nlan   = lane & 15;
    const int m0 = wave * 16;          // 8 waves cover B=128
    const int n0 = blockIdx.x * 16;    // 32 blocks cover H=512

    // --- cooperative stage of Wh tile: rows (g*HID + n0 + 0..15), g=0..2 ---
    for (int i = tid; i < 48 * 64; i += 256) {    // 64 x 16B chunks per row
        const int row   = i >> 6;                 // 0..47
        const int c     = (i & 63) * 8;
        const int gate  = row >> 4;
        const int rowIn = row & 15;
        stage16(Whbf + (size_t)(gate * HID + n0 + rowIn) * HID + c,
                smem + row * CELL_LDS_RS + c);
    }
    stage_fence();

    v8f aR = {}, aI = {}, aN = {};
    const __bf16* arow = hbfIn + (size_t)(m0 + nlan) * HID;
    const __bf16* bRr = smem + ( 0 + nlan) * CELL_LDS_RS;
    const __bf16* bIr = smem + (16 + nlan) * CELL_LDS_RS;
    const __bf16* bNr = smem + (32 + nlan) * CELL_LDS_RS;

#pragma unroll 4
    for (int k0 = 0; k0 < HID; k0 += 32) {
        v16bf a  = load_a_frag(arow, k0, halfId);
        v16bf bR = load_b_row(bRr, k0, halfId);
        v16bf bI = load_b_row(bIr, k0, halfId);
        v16bf bN = load_b_row(bNr, k0, halfId);
        aR = wmma_bf16(a, bR, aR);
        aI = wmma_bf16(a, bI, aI);
        aN = wmma_bf16(a, bN, aN);
    }

    const float sb  = sigmoidf_(betaPtr[0]);
    const float b21 = 1.0f / (2.0f * sb);
    const int   ncol = n0 + nlan;
    const float bhr = bh[0 * HID + ncol];
    const float bhi = bh[1 * HID + ncol];
    const float bhn = bh[2 * HID + ncol];

#pragma unroll
    for (int r = 0; r < 8; ++r) {
        const int mrow = m0 + r + 8 * halfId;
        const size_t ix = (size_t)mrow * HID + ncol;
        const float* gxr = gxs + (size_t)mrow * G3;

        const float ir  = gxr[0 * HID + ncol];
        const float ii  = gxr[1 * HID + ncol];
        const float inn = gxr[2 * HID + ncol];
        const float hr  = aR[r] + bhr;
        const float hi  = aI[r] + bhi;
        const float hnn = aN[r] + bhn;

        const float rg = sigmoidf_(ir + hr);
        const float ig = sigmoidf_(ii + hi);
        const float ng = tanhf(inn + rg * hnn);
        const float hval = hcur[ix];

        if (phase == 0) {
            const float f = (1.0f - ig) * (ng - hval);
            f1[ix]    = f;
            bfOut[ix] = (__bf16)(hval + sb * f);
        } else {
            const float f1v   = f1[ix];
            const float halfv = hval + sb * f1v;
            const float f2    = (1.0f - ig) * (ng - halfv);
            const float hn    = hval + (1.0f - b21) * f1v + b21 * f2;
            fOut[ix]  = hn;
            bfOut[ix] = (__bf16)hn;
            if (outF32) outF32[ix] = hn;
        }
    }
}

// ---------------------------------------------------------------------------
// Host-side orchestration
// ---------------------------------------------------------------------------
extern "C" void kernel_launch(void* const* d_in, const int* in_sizes, int n_in,
                              void* d_out, int out_size, void* d_ws, size_t ws_size,
                              hipStream_t stream) {
    (void)in_sizes; (void)n_in; (void)out_size; (void)ws_size;

    const float* x     = (const float*)d_in[0];   // [S,B,I]
    const float* Wx0   = (const float*)d_in[1];   // [3H,I]
    const float* bx0   = (const float*)d_in[2];
    const float* Wh0   = (const float*)d_in[3];   // [3H,H]
    const float* bh0   = (const float*)d_in[4];
    const float* beta0 = (const float*)d_in[5];
    const float* Wx1   = (const float*)d_in[6];   // [3H,H]
    const float* bx1   = (const float*)d_in[7];
    const float* Wh1   = (const float*)d_in[8];
    const float* bh1   = (const float*)d_in[9];
    const float* beta1 = (const float*)d_in[10];
    float* dout = (float*)d_out;                  // [S,B,H] + [B,H]

    // --- carve workspace (256B aligned) ---
    char* p = (char*)d_ws;
    auto carve = [&](size_t bytes) -> void* {
        void* r = (void*)p;
        p += (bytes + 255) & ~(size_t)255;
        return r;
    };
    const size_t BH = (size_t)BATCH * HID;
    __bf16* xbf     = (__bf16*)carve((size_t)S_LEN * BATCH * IN0 * 2);
    __bf16* x1bf    = (__bf16*)carve((size_t)S_LEN * BH * 2);   // layer-0 out / layer-1 in
    __bf16* Wx0bf   = (__bf16*)carve((size_t)G3 * IN0 * 2);
    __bf16* Wh0bf   = (__bf16*)carve((size_t)G3 * HID * 2);
    __bf16* Wx1bf   = (__bf16*)carve((size_t)G3 * HID * 2);
    __bf16* Wh1bf   = (__bf16*)carve((size_t)G3 * HID * 2);
    float*  gxc     = (float*)carve((size_t)CHUNK * BATCH * G3 * 4);
    float*  hf0     = (float*)carve(BH * 4);
    float*  hf1     = (float*)carve(BH * 4);
    float*  f1buf   = (float*)carve(BH * 4);
    __bf16* hbf0    = (__bf16*)carve(BH * 2);
    __bf16* hbf1    = (__bf16*)carve(BH * 2);
    __bf16* halfbf  = (__bf16*)carve(BH * 2);
    __bf16* hbfZero = (__bf16*)carve(BH * 2);
    float*  hfbuf[2]  = {hf0, hf1};
    __bf16* hbfbuf[2] = {hbf0, hbf1};

    auto grid1d = [](size_t n) { return (unsigned)((n + 255) / 256); };

    // --- bf16 copies of GEMM operands ---
    {
        size_t n = (size_t)S_LEN * BATCH * IN0;
        k_f2bf<<<grid1d(n), 256, 0, stream>>>(x, xbf, n);
        n = (size_t)G3 * IN0;
        k_f2bf<<<grid1d(n), 256, 0, stream>>>(Wx0, Wx0bf, n);
        n = (size_t)G3 * HID;
        k_f2bf<<<grid1d(n), 256, 0, stream>>>(Wh0, Wh0bf, n);
        k_f2bf<<<grid1d(n), 256, 0, stream>>>(Wx1, Wx1bf, n);
        k_f2bf<<<grid1d(n), 256, 0, stream>>>(Wh1, Wh1bf, n);
    }

    const int Mc = CHUNK * BATCH;                            // 8192 rows per chunk
    const unsigned gemmBlocks = (unsigned)((Mc / 128) * (G3 / 64));   // 64*24 = 1536

    // =====================  Layer 0  =====================
    k_zero_f32<<<grid1d(BH), 256, 0, stream>>>(hfbuf[0], BH);
    k_zero_bf <<<grid1d(BH), 256, 0, stream>>>(hbfZero, BH);

    for (int base = 0; base < S_LEN; base += CHUNK) {
        k_gemm_bias<<<gemmBlocks, 256, 0, stream>>>(
            xbf + (size_t)base * BATCH * IN0, Wx0bf, bx0, gxc, Mc, G3, IN0, /*log2(256/8)=*/5);
        for (int sl = 0; sl < CHUNK; ++sl) {
            const int s = base + sl;
            const float*  gxs   = gxc + (size_t)sl * BATCH * G3;
            const __bf16* hbfIn = (s == 0) ? hbfZero : (x1bf + (size_t)(s - 1) * BH);
            const float*  hcur  = hfbuf[s & 1];
            k_cell<<<32, 256, 0, stream>>>(hbfIn, Wh0bf, bh0, gxs, hcur, beta0,
                                           f1buf, halfbf, nullptr, nullptr, 0);
            k_cell<<<32, 256, 0, stream>>>(halfbf, Wh0bf, bh0, gxs, hcur, beta0,
                                           f1buf, x1bf + (size_t)s * BH,
                                           hfbuf[(s + 1) & 1], nullptr, 1);
        }
    }

    // =====================  Layer 1  =====================
    k_zero_f32<<<grid1d(BH), 256, 0, stream>>>(hfbuf[0], BH);
    k_zero_bf <<<grid1d(BH), 256, 0, stream>>>(hbfbuf[0], BH);

    for (int base = 0; base < S_LEN; base += CHUNK) {
        k_gemm_bias<<<gemmBlocks, 256, 0, stream>>>(
            x1bf + (size_t)base * BH, Wx1bf, bx1, gxc, Mc, G3, HID, /*log2(512/8)=*/6);
        for (int sl = 0; sl < CHUNK; ++sl) {
            const int s = base + sl;
            const float* gxs  = gxc + (size_t)sl * BATCH * G3;
            const float* hcur = hfbuf[s & 1];
            k_cell<<<32, 256, 0, stream>>>(hbfbuf[s & 1], Wh1bf, bh1, gxs, hcur, beta1,
                                           f1buf, halfbf, nullptr, nullptr, 0);
            k_cell<<<32, 256, 0, stream>>>(halfbf, Wh1bf, bh1, gxs, hcur, beta1,
                                           f1buf, hbfbuf[(s + 1) & 1],
                                           hfbuf[(s + 1) & 1],
                                           dout + (size_t)s * BH, 1);
        }
    }

    // append h2[-1]
    k_copy_f32<<<grid1d(BH), 256, 0, stream>>>(
        dout + (size_t)(S_LEN - 1) * BH, dout + (size_t)S_LEN * BH, BH);
}
